// GatedDeltaNetVarlen_88330297410071
// MI455X (gfx1250) — matrix operand order repaired and checked by reading, not compiled, not run on hardware
//
#include <hip/hip_runtime.h>

// ---------------------------------------------------------------------------
// Types / bf16 helpers
// ---------------------------------------------------------------------------
typedef __attribute__((ext_vector_type(16))) __bf16 v16bf;
typedef __attribute__((ext_vector_type(8)))  float  v8f;

__device__ __forceinline__ float bf2f(unsigned short u) {
    unsigned int x = ((unsigned int)u) << 16;
    return __builtin_bit_cast(float, x);
}
__device__ __forceinline__ unsigned short f2bfbits(float f) {
    unsigned int x = __builtin_bit_cast(unsigned int, f);
    unsigned int lsb = (x >> 16) & 1u;
    x += 0x7fffu + lsb;                 // round-to-nearest-even
    return (unsigned short)(x >> 16);
}
__device__ __forceinline__ __bf16 bf_from_bits(unsigned short u) {
    return __builtin_bit_cast(__bf16, u);
}
__device__ __forceinline__ __bf16 f2bf(float f) { return bf_from_bits(f2bfbits(f)); }
__device__ __forceinline__ float siluf(float x) { return x / (1.f + expf(-x)); }

// ---------------------------------------------------------------------------
// Problem constants (match reference setup_inputs)
// ---------------------------------------------------------------------------
static constexpr int Bb      = 4;
static constexpr int Ll      = 2048;
static constexpr int Ttok    = Bb * Ll;        // 8192
static constexpr int HID     = 2048;
static constexpr int HKc     = 16;
static constexpr int DKc     = 128;
static constexpr int HVc     = 32;
static constexpr int DVc     = 128;
static constexpr int KEYDIM  = HKc * DKc;      // 2048
static constexpr int VALDIM  = HVc * DVc;      // 4096
static constexpr int CONVDIM = 2 * KEYDIM + VALDIM;  // 8192
static constexpr int KW      = 4;
static constexpr int CC      = 64;             // chunk size
static constexpr int NCH     = Ll / CC;        // 32

// ---------------------------------------------------------------------------
// Fragment loaders for the core micro-GEMM (all contiguous-u16 layouts).
// ---------------------------------------------------------------------------
struct LA_RM   { const unsigned short* p; int ld;
    __device__ __bf16 operator()(int m, int k) const { return bf_from_bits(p[m * ld + k]); } };
struct LA_RM_MS{ const unsigned short* p; int ld; const float* s;   // scale by s[m]
    __device__ __bf16 operator()(int m, int k) const { return f2bf(bf2f(p[m * ld + k]) * s[m]); } };
struct LA_RM_KS{ const unsigned short* p; int ld; const float* s;   // scale by s[k]
    __device__ __bf16 operator()(int m, int k) const { return f2bf(bf2f(p[m * ld + k]) * s[k]); } };
struct LB_NM   { const unsigned short* p; int ld;                   // p[n*ld + k]
    __device__ __bf16 operator()(int k, int n) const { return bf_from_bits(p[n * ld + k]); } };

// mm_lds operand layouts (contiguous k order chosen here, not the ISA table order):
//  A 16x32 bf16 : lane = 16*half + row ; elem e -> k = half*16 + e
//  B 32x16 bf16 : lane = 16*half + col ; elem e -> k = half*16 + e
//  C 16x16 f32  : vgpr r -> m = r + half*8 ; n = lane&15
template <int M, int N, int K, int NW, class FA, class FB>
__device__ void mm_lds(FA fa, FB fb, float* C, int ldc, float alpha, float beta, int tid) {
    const int lane  = tid & 31;
    const int w     = tid >> 5;
    const int row16 = lane & 15;
    const int half  = lane >> 4;
    constexpr int NT = (M / 16) * (N / 16);
    for (int t = w; t < NT; t += NW) {
        const int m0 = (t / (N / 16)) * 16;
        const int n0 = (t % (N / 16)) * 16;
        v8f acc = {};
        for (int k0 = 0; k0 < K; k0 += 32) {
            v16bf af, bfv;
#pragma unroll
            for (int e = 0; e < 16; ++e) {
                const int ka = half * 16 + e;
                af[e]  = fa(m0 + row16, k0 + ka);
                const int kb = half * 16 + e;
                bfv[e] = fb(k0 + kb, n0 + row16);
            }
            acc = __builtin_amdgcn_wmma_f32_16x16x32_bf16(false, af, false, bfv,
                                                          (short)0, acc, false, false);
        }
#pragma unroll
        for (int r = 0; r < 8; ++r) {
            const int m = m0 + r + half * 8;
            const int n = n0 + row16;
            float old = (beta != 0.f) ? C[m * ldc + n] : 0.f;
            C[m * ldc + n] = alpha * acc[r] + beta * old;
        }
    }
}

// ---------------------------------------------------------------------------
// Kernel: f32 -> bf16 conversion (plain)
// ---------------------------------------------------------------------------
__global__ void f32_to_bf16_kernel(const float* __restrict__ src,
                                   unsigned short* __restrict__ dst, long n) {
    long i = (long)blockIdx.x * blockDim.x + threadIdx.x;
    if (i < n) dst[i] = f2bfbits(src[i]);
}

// ---------------------------------------------------------------------------
// Kernel: f32 (R x C) -> bf16 transposed (C x R), tiled through LDS
// ---------------------------------------------------------------------------
__global__ __launch_bounds__(256)
void f32_to_bf16_t_kernel(const float* __restrict__ src,
                          unsigned short* __restrict__ dst, int R, int Cn) {
    __shared__ float tile[32][33];
    const int bx = blockIdx.x * 32;   // col base in src
    const int by = blockIdx.y * 32;   // row base in src
    const int tx = threadIdx.x & 31, ty = threadIdx.x >> 5;
    for (int j = ty; j < 32; j += 8)
        tile[j][tx] = src[(size_t)(by + j) * Cn + bx + tx];
    __syncthreads();
    for (int j = ty; j < 32; j += 8)
        dst[(size_t)(bx + j) * R + by + tx] = f2bfbits(tile[tx][j]);
}

// ---------------------------------------------------------------------------
// Kernel: big bf16 WMMA GEMM   C(f32, MxN) = A(bf16, MxK) @ Bt(bf16, NxK)^T
// 128x128x32 tiling, double-buffered LDS: tile k+1 prefetched to registers
// before the WMMA block of tile k, stored to alternate buffer after compute,
// one barrier per K step. Fragments are ds_load_b128 pairs (padded LDT rows).
// ---------------------------------------------------------------------------
#define GBM 128
#define GBN 128
#define GBK 32
#define LDT 48
__global__ __launch_bounds__(256)
void gemm_bf16_tn_kernel(const unsigned short* __restrict__ A,
                         const unsigned short* __restrict__ Bt,
                         float* __restrict__ C, int Mdim, int Ndim, int Kdim) {
    __shared__ __align__(16) unsigned short As[2][GBM * LDT];
    __shared__ __align__(16) unsigned short Bs[2][GBN * LDT];
    const int tid  = threadIdx.x;
    const int lane = tid & 31, w = tid >> 5;
    const int bm = blockIdx.y * GBM, bn = blockIdx.x * GBN;
    const int wr = w >> 2, wc = w & 3;              // 2 x 4 wave grid
    const int row16 = lane & 15, half = lane >> 4;

    // Per-thread staging slots: rows r0 and r0+64 at column c0 (uint4 = 8 ush).
    const int r0 = tid >> 2, c0 = (tid & 3) * 8;
    const unsigned short* pA0 = A  + (size_t)(bm + r0) * Kdim + c0;
    const unsigned short* pA1 = pA0 + (size_t)64 * Kdim;
    const unsigned short* pB0 = Bt + (size_t)(bn + r0) * Kdim + c0;
    const unsigned short* pB1 = pB0 + (size_t)64 * Kdim;
    const int s0 = r0 * LDT + c0, s1 = (r0 + 64) * LDT + c0;

    v8f acc[4][2] = {};

    // prologue: tile 0 -> buffer 0
    *(uint4*)(As[0] + s0) = *(const uint4*)(pA0);
    *(uint4*)(As[0] + s1) = *(const uint4*)(pA1);
    *(uint4*)(Bs[0] + s0) = *(const uint4*)(pB0);
    *(uint4*)(Bs[0] + s1) = *(const uint4*)(pB1);
    __syncthreads();

    const int ntile = Kdim / GBK;
    for (int it = 0; it < ntile; ++it) {
        uint4 ra0, ra1, rb0, rb1;
        const bool pre = (it + 1) < ntile;
        if (pre) {
            const int k = (it + 1) * GBK;
            ra0 = *(const uint4*)(pA0 + k);
            ra1 = *(const uint4*)(pA1 + k);
            rb0 = *(const uint4*)(pB0 + k);
            rb1 = *(const uint4*)(pB1 + k);
        }
        const unsigned short* as = As[it & 1];
        const unsigned short* bs = Bs[it & 1];
        v16bf bfr[2];
#pragma unroll
        for (int ns = 0; ns < 2; ++ns) {
            const int ncol = wc * 32 + ns * 16 + row16;
#pragma unroll
            for (int e = 0; e < 16; ++e)
                bfr[ns][e] = bf_from_bits(bs[ncol * LDT + half * 16 + e]);
        }
#pragma unroll
        for (int ms = 0; ms < 4; ++ms) {
            v16bf afr;
            const int arow = wr * 64 + ms * 16 + row16;
#pragma unroll
            for (int e = 0; e < 16; ++e) {
                const int kk = half * 16 + e;
                afr[e] = bf_from_bits(as[arow * LDT + kk]);
            }
#pragma unroll
            for (int ns = 0; ns < 2; ++ns)
                acc[ms][ns] = __builtin_amdgcn_wmma_f32_16x16x32_bf16(
                    false, afr, false, bfr[ns], (short)0, acc[ms][ns], false, false);
        }
        if (pre) {
            unsigned short* ad = As[(it + 1) & 1];
            unsigned short* bd = Bs[(it + 1) & 1];
            *(uint4*)(ad + s0) = ra0;
            *(uint4*)(ad + s1) = ra1;
            *(uint4*)(bd + s0) = rb0;
            *(uint4*)(bd + s1) = rb1;
        }
        __syncthreads();
    }
    const size_t cbase = (size_t)(bm + wr * 64 + half * 8) * Ndim
                       + (size_t)(bn + wc * 32 + row16);
#pragma unroll
    for (int ms = 0; ms < 4; ++ms)
#pragma unroll
        for (int ns = 0; ns < 2; ++ns)
#pragma unroll
            for (int r = 0; r < 8; ++r)
                C[cbase + (size_t)(ms * 16 + r) * Ndim + ns * 16] = acc[ms][ns][r];
}

// ---------------------------------------------------------------------------
// Kernel: b/a projections + beta/g nonlinearities
// ---------------------------------------------------------------------------
__global__ __launch_bounds__(256)
void ba_kernel(const float* __restrict__ hs, const float* __restrict__ Wb,
               const float* __restrict__ Wa, const float* __restrict__ dt_bias,
               const float* __restrict__ A_log, float* __restrict__ beta,
               float* __restrict__ g) {
    __shared__ float row[HID];
    const long t = blockIdx.x;
    for (int i = threadIdx.x; i < HID; i += 256) row[i] = hs[t * HID + i];
    __syncthreads();
    const int tid = threadIdx.x;
    if (tid < 2 * HVc) {
        const int h = tid & (HVc - 1);
        const bool isA = tid >= HVc;
        const float* W = isA ? Wa : Wb;
        float acc = 0.f;
        for (int i = 0; i < HID; ++i) acc += row[i] * W[i * HVc + h];
        if (isA) {
            const float x  = acc + dt_bias[h];
            const float sp = (x > 20.f) ? x : log1pf(expf(x));
            g[t * HVc + h] = -expf(A_log[h]) * sp;
        } else {
            beta[t * HVc + h] = 1.f / (1.f + expf(-acc));
        }
    }
}

// ---------------------------------------------------------------------------
// Kernel: causal depthwise conv1d + SiLU. 4 outputs per thread (7 loads).
// ---------------------------------------------------------------------------
__global__ void conv_silu_kernel(const float* __restrict__ mixed,
                                 const float* __restrict__ cw,
                                 float* __restrict__ out) {
    const long idx = (long)blockIdx.x * blockDim.x + threadIdx.x;
    const long total = (long)Ttok * CONVDIM / 4;
    if (idx >= total) return;
    const int  c  = (int)(idx % CONVDIM);
    const long t0 = (idx / CONVDIM) * 4;
    const int  l0 = (int)(t0 % Ll);
    const float w0 = cw[c * KW], w1 = cw[c * KW + 1], w2 = cw[c * KW + 2], w3 = cw[c * KW + 3];
    float x[7];
#pragma unroll
    for (int j = 0; j < 7; ++j) {
        const int lj = l0 - 3 + j;
        x[j] = (lj >= 0) ? mixed[(t0 - 3 + j) * CONVDIM + c] : 0.f;
    }
#pragma unroll
    for (int u = 0; u < 4; ++u)
        out[(t0 + u) * CONVDIM + c] =
            siluf(x[u] * w0 + x[u + 1] * w1 + x[u + 2] * w2 + x[u + 3] * w3);
}

// ---------------------------------------------------------------------------
// Kernel: chunked gated delta rule core. One workgroup (16 waves / 512 thr)
// per (b,h). S (128x128 f32) + bf16 operand tiles resident in LDS (~323 KB).
// ---------------------------------------------------------------------------
static constexpr int  CORE_THREADS = 512;
static constexpr int  CORE_NW      = CORE_THREADS / 32;   // 16 waves
static constexpr size_t CORE_F32 =
    128 * 128 + 64 * 128 * 2 + 64 * 64 * 2 + 64 * 4 + 512 + 8;      // S,Wt,U,M64,Tm,small
static constexpr size_t CORE_U16 =
    64 * 128 * 4 + 128 * 128 + 64 * 128 * 2 + 64 * 64 * 3;          // q,k,kt,v | Sbt | vn,kc | Tb,Tbe,attn
static constexpr size_t CORE_SMEM_BYTES = CORE_F32 * 4 + CORE_U16 * 2;

__global__ __launch_bounds__(CORE_THREADS)
void core_kernel(const float* __restrict__ convs, const float* __restrict__ gArr,
                 const float* __restrict__ betaArr, float* __restrict__ coreOut) {
    extern __shared__ char smem[];
    float* S     = (float*)smem;           // 128x128 state (dk x dv)
    float* Wt    = S + 128 * 128;          // 64x128  v2 -> v_new (f32)
    float* U     = Wt + 64 * 128;          // 64x128  kcum -> out (f32)
    float* M64   = U + 64 * 128;           // 64x64   P/A (f32)
    float* Tm    = M64 + 64 * 64;          // 64x64   (I-A)^-1
    float* gc    = Tm + 64 * 64;
    float* betaC = gc + 64;
    float* egc   = betaC + 64;
    float* sexp  = egc + 64;               // exp(gl - gc)
    float* rn8   = sexp + 64;              // 64x8 norm partials
    float* scal  = rn8 + 512;              // [0]=gl [1]=e^gl
    unsigned short* qbf   = (unsigned short*)(scal + 8);  // 64x128 row-major
    unsigned short* kbf   = qbf + 64 * 128;               // 64x128 row-major
    unsigned short* kbt   = kbf + 64 * 128;               // 128x64 (d-major)
    unsigned short* vbt   = kbt + 64 * 128;               // 128x64 (d-major)
    unsigned short* Sbt   = vbt + 64 * 128;               // 128x128 (dv-major)
    unsigned short* vnbt  = Sbt + 128 * 128;              // 128x64 (dv-major) v_new^T
    unsigned short* kcb   = vnbt + 64 * 128;              // 64x128 row-major kcum
    unsigned short* Tbbf  = kcb + 64 * 128;               // 64x64
    unsigned short* Tbebf = Tbbf + 64 * 64;               // 64x64
    unsigned short* attnbf= Tbebf + 64 * 64;              // 64x64

    const int tid = threadIdx.x;
    const int bh  = blockIdx.x;
    const int b   = bh / HVc;
    const int h   = bh % HVc;
    const int hk  = h >> 1;                // rep = Hv/HK = 2

    for (int i = tid; i < 128 * 128; i += CORE_THREADS) S[i] = 0.f;
    __syncthreads();

    const int rr = tid >> 3, pp = tid & 7; // row / eighth (16 elems) for normalization

    for (int n = 0; n < NCH; ++n) {
        const long tok0 = (long)b * Ll + (long)n * CC;

        // ---- Sbt = bf16(S^T), from previous chunk's state ----
        for (int i = tid; i < 128 * 128; i += CORE_THREADS) {
            const int dv = i >> 7, dk = i & 127;
            Sbt[i] = f2bfbits(S[dk * 128 + dv]);
        }
        // ---- q : registers -> l2norm -> * Dk^-0.5 -> qbf ----
        {
            float v[16]; float s = 0.f;
            const float* src = convs + (tok0 + rr) * CONVDIM + hk * 128 + pp * 16;
#pragma unroll
            for (int j = 0; j < 16; ++j) { v[j] = src[j]; s += v[j] * v[j]; }
            rn8[rr * 8 + pp] = s;
            __syncthreads();
            float nrm = 0.f;
#pragma unroll
            for (int q8 = 0; q8 < 8; ++q8) nrm += rn8[rr * 8 + q8];
            const float sc = rsqrtf(nrm + 1e-6f) * 0.08838834764831845f;
#pragma unroll
            for (int j = 0; j < 16; ++j) qbf[rr * 128 + pp * 16 + j] = f2bfbits(v[j] * sc);
        }
        __syncthreads();
        // ---- k : registers -> l2norm -> kbf (row-major) + kbt (d-major) ----
        {
            float v[16]; float s = 0.f;
            const float* src = convs + (tok0 + rr) * CONVDIM + KEYDIM + hk * 128 + pp * 16;
#pragma unroll
            for (int j = 0; j < 16; ++j) { v[j] = src[j]; s += v[j] * v[j]; }
            rn8[rr * 8 + pp] = s;
            __syncthreads();
            float nrm = 0.f;
#pragma unroll
            for (int q8 = 0; q8 < 8; ++q8) nrm += rn8[rr * 8 + q8];
            const float sc = rsqrtf(nrm + 1e-6f);
#pragma unroll
            for (int j = 0; j < 16; ++j) {
                const unsigned short ub = f2bfbits(v[j] * sc);
                kbf[rr * 128 + pp * 16 + j] = ub;
                kbt[(pp * 16 + j) * 64 + rr] = ub;
            }
        }
        // ---- v -> vbt (d-major) ----
        for (int i = tid; i < 64 * 128; i += CORE_THREADS) {
            const int t = i & 63, d = i >> 6;
            vbt[i] = f2bfbits(convs[(tok0 + t) * CONVDIM + 2 * KEYDIM + h * 128 + d]);
        }
        // ---- g cumsum, beta ----
        if (tid < 64) {
            gc[tid]    = gArr[(tok0 + tid) * HVc + h];
            betaC[tid] = betaArr[(tok0 + tid) * HVc + h];
        }
        __syncthreads();
        if (tid == 0) {
            float a = 0.f;
            for (int i = 0; i < 64; ++i) { a += gc[i]; gc[i] = a; }
            scal[0] = a; scal[1] = expf(a);
        }
        __syncthreads();
        if (tid < 64) { egc[tid] = expf(gc[tid]); sexp[tid] = expf(scal[0] - gc[tid]); }
        __syncthreads();

        // P = k @ k^T
        mm_lds<64, 64, 128, CORE_NW>(LA_RM{kbf, 128}, LB_NM{kbf, 128}, M64, 64, 1.f, 0.f, tid);
        __syncthreads();
        // A = -(beta_i * P * decay), strictly lower (f32, for substitution)
        for (int i = tid; i < 4096; i += CORE_THREADS) {
            const int r = i >> 6, c = i & 63;
            M64[i] = (r > c) ? -betaC[r] * M64[i] * expf(gc[r] - gc[c]) : 0.f;
        }
        __syncthreads();
        // T = (I - A)^{-1}: forward substitution, one column per lane (no races)
        if (tid < 64) {
            for (int i2 = 0; i2 < 64; ++i2) {
                float s = (i2 == tid) ? 1.f : 0.f;
                for (int m = 0; m < i2; ++m) s += M64[i2 * 64 + m] * Tm[m * 64 + tid];
                Tm[i2 * 64 + tid] = s;
            }
        }
        __syncthreads();
        // Tb = T*diag(beta), Tbe = T*diag(beta*e^gc)
        for (int i = tid; i < 4096; i += CORE_THREADS) {
            const int m = i & 63;
            const float v = Tm[i] * betaC[m];
            Tbbf[i]  = f2bfbits(v);
            Tbebf[i] = f2bfbits(v * egc[m]);
        }
        __syncthreads();
        // v2 = Tb @ v ; kcum = Tbe @ k ; attnP = q @ k^T
        mm_lds<64, 128, 64, CORE_NW>(LA_RM{Tbbf, 64},  LB_NM{vbt, 64}, Wt, 128, 1.f, 0.f, tid);
        mm_lds<64, 128, 64, CORE_NW>(LA_RM{Tbebf, 64}, LB_NM{kbt, 64}, U,  128, 1.f, 0.f, tid);
        mm_lds<64, 64, 128, CORE_NW>(LA_RM{qbf, 128},  LB_NM{kbf, 128}, M64, 64, 1.f, 0.f, tid);
        __syncthreads();
        // kcum -> bf16 row-major ; attn mask -> bf16
        for (int i = tid; i < 64 * 128; i += CORE_THREADS) kcb[i] = f2bfbits(U[i]);
        for (int i = tid; i < 4096; i += CORE_THREADS) {
            const int r = i >> 6, c = i & 63;
            attnbf[i] = f2bfbits((r >= c) ? M64[i] * expf(gc[r] - gc[c]) : 0.f);
        }
        __syncthreads();
        // v_new = v2 - kcum @ S
        mm_lds<64, 128, 128, CORE_NW>(LA_RM{kcb, 128}, LB_NM{Sbt, 128}, Wt, 128, -1.f, 1.f, tid);
        __syncthreads();
        // v_new -> bf16 dv-major
        for (int i = tid; i < 64 * 128; i += CORE_THREADS) {
            const int t = i & 63, dv = i >> 6;
            vnbt[i] = f2bfbits(Wt[t * 128 + dv]);
        }
        __syncthreads();
        // out = (q * e^gc) @ S + attn @ v_new
        mm_lds<64, 128, 128, CORE_NW>(LA_RM_MS{qbf, 128, egc}, LB_NM{Sbt, 128}, U, 128, 1.f, 0.f, tid);
        __syncthreads();
        mm_lds<64, 128, 64, CORE_NW>(LA_RM{attnbf, 64}, LB_NM{vnbt, 64}, U, 128, 1.f, 1.f, tid);
        __syncthreads();
        for (int i = tid; i < 64 * 128; i += CORE_THREADS) {
            const int r = i >> 7, d = i & 127;
            coreOut[(tok0 + r) * (long)VALDIM + h * 128 + d] = U[i];
        }
        // S = S*e^gl + (k*e^{gl-gc})^T @ v_new
        const float egl = scal[1];
        for (int i = tid; i < 128 * 128; i += CORE_THREADS) S[i] *= egl;
        __syncthreads();
        mm_lds<128, 128, 64, CORE_NW>(LA_RM_KS{kbt, 64, sexp}, LB_NM{vnbt, 64}, S, 128, 1.f, 1.f, tid);
        __syncthreads();
    }
}

// ---------------------------------------------------------------------------
// Kernel: gated RMSNorm -> bf16 y. One wave32 per 128-wide row.
// ---------------------------------------------------------------------------
__global__ __launch_bounds__(256)
void gate_norm_kernel(const float* __restrict__ core, const float* __restrict__ z,
                      const float* __restrict__ norm_w, unsigned short* __restrict__ ybf) {
    const int lane = threadIdx.x & 31, w = threadIdx.x >> 5;
    const long r = (long)blockIdx.x * 8 + w;
    const float* cp = core + r * 128;
    const float* zp = z + r * 128;
    float vals[4];
    float ss = 0.f;
#pragma unroll
    for (int i = 0; i < 4; ++i) {
        const int d = lane + 32 * i;
        const float gv = cp[d] * siluf(zp[d]);
        vals[i] = gv; ss += gv * gv;
    }
#pragma unroll
    for (int o = 16; o > 0; o >>= 1) ss += __shfl_xor(ss, o, 32);
    const float rs = rsqrtf(ss * (1.f / 128.f) + 1e-6f);
#pragma unroll
    for (int i = 0; i < 4; ++i) {
        const int d = lane + 32 * i;
        ybf[r * 128 + d] = f2bfbits(vals[i] * rs * norm_w[d]);
    }
}

// ---------------------------------------------------------------------------
// Host launcher
// ---------------------------------------------------------------------------
extern "C" void kernel_launch(void* const* d_in, const int* in_sizes, int n_in,
                              void* d_out, int out_size, void* d_ws, size_t ws_size,
                              hipStream_t stream) {
    const float* hidden  = (const float*)d_in[0];
    const float* W_qkv   = (const float*)d_in[1];
    const float* W_z     = (const float*)d_in[2];
    const float* W_b     = (const float*)d_in[3];
    const float* W_a     = (const float*)d_in[4];
    const float* conv_w  = (const float*)d_in[5];
    const float* dt_bias = (const float*)d_in[6];
    const float* A_log   = (const float*)d_in[7];
    const float* norm_w  = (const float*)d_in[8];
    const float* W_out   = (const float*)d_in[9];
    float* out = (float*)d_out;

    char* ws = (char*)d_ws;
    auto carve = [&](size_t bytes) {
        char* p = ws;
        ws += (bytes + 255) & ~(size_t)255;
        return p;
    };
    unsigned short* hbf      = (unsigned short*)carve((size_t)Ttok * HID * 2);
    unsigned short* wqkv_bt  = (unsigned short*)carve((size_t)HID * CONVDIM * 2);   // CONVDIM x HID
    unsigned short* wz_bt    = (unsigned short*)carve((size_t)HID * VALDIM * 2);    // VALDIM  x HID
    unsigned short* wout_bt  = (unsigned short*)carve((size_t)VALDIM * HID * 2);    // HID x VALDIM
    float* mixed             = (float*)carve((size_t)Ttok * CONVDIM * 4);  // reused as coreOut
    float* convs             = (float*)carve((size_t)Ttok * CONVDIM * 4);  // reused as ybf
    float* zbuf              = (float*)carve((size_t)Ttok * VALDIM * 4);
    float* betaBuf           = (float*)carve((size_t)Ttok * HVc * 4);
    float* gBuf              = (float*)carve((size_t)Ttok * HVc * 4);
    float* coreOut           = mixed;
    unsigned short* ybf      = (unsigned short*)convs;

    // 1. Convert activations (plain) and weights (transposed) to bf16
    {
        long n = (long)Ttok * HID;
        f32_to_bf16_kernel<<<(unsigned)((n + 255) / 256), 256, 0, stream>>>(hidden, hbf, n);
        dim3 g1(CONVDIM / 32, HID / 32);
        f32_to_bf16_t_kernel<<<g1, 256, 0, stream>>>(W_qkv, wqkv_bt, HID, CONVDIM);
        dim3 g2(VALDIM / 32, HID / 32);
        f32_to_bf16_t_kernel<<<g2, 256, 0, stream>>>(W_z, wz_bt, HID, VALDIM);
        dim3 g3(HID / 32, VALDIM / 32);
        f32_to_bf16_t_kernel<<<g3, 256, 0, stream>>>(W_out, wout_bt, VALDIM, HID);
    }
    // 2. mixed = hidden @ W_qkv   (8192 x 8192 x 2048)
    {
        dim3 grid(CONVDIM / GBN, Ttok / GBM);
        gemm_bf16_tn_kernel<<<grid, 256, 0, stream>>>(hbf, wqkv_bt, mixed, Ttok, CONVDIM, HID);
    }
    // 3. z = hidden @ W_z
    {
        dim3 grid(VALDIM / GBN, Ttok / GBM);
        gemm_bf16_tn_kernel<<<grid, 256, 0, stream>>>(hbf, wz_bt, zbuf, Ttok, VALDIM, HID);
    }
    // 4. beta / g
    ba_kernel<<<Ttok, 256, 0, stream>>>(hidden, W_b, W_a, dt_bias, A_log, betaBuf, gBuf);
    // 5. causal conv + SiLU
    {
        long n = (long)Ttok * CONVDIM / 4;
        conv_silu_kernel<<<(unsigned)((n + 255) / 256), 256, 0, stream>>>(mixed, conv_w, convs);
    }
    // 6. delta-rule core (dynamic LDS ~323 KB, within 320 KB/WGP arch limit)
    {
        hipFuncSetAttribute((const void*)core_kernel,
                            hipFuncAttributeMaxDynamicSharedMemorySize,
                            (int)CORE_SMEM_BYTES);
        core_kernel<<<Bb * HVc, CORE_THREADS, CORE_SMEM_BYTES, stream>>>(convs, gBuf, betaBuf, coreOut);
    }
    // 7. gated RMSNorm -> bf16 y
    {
        const long rows = (long)Ttok * HVc;
        gate_norm_kernel<<<(unsigned)(rows / 8), 256, 0, stream>>>(coreOut, zbuf, norm_w, ybf);
    }
    // 8. out = y @ W_out   (8192 x 2048 x 4096)
    {
        dim3 grid(HID / GBN, Ttok / GBM);
        gemm_bf16_tn_kernel<<<grid, 256, 0, stream>>>(ybf, wout_bt, out, Ttok, HID, VALDIM);
    }
    (void)in_sizes; (void)n_in; (void)out_size; (void)ws_size;
}
